// MoE_84619445666065
// MI455X (gfx1250) — matrix operand order repaired and checked by reading, not compiled, not run on hardware
//
#include <hip/hip_runtime.h>

// ---- problem dims (match reference) ----
#define BTOK 4096
#define DDIM 1024
#define HDIM 4096
#define ODIM 1024
#define NEXP 8
#define NACT 5

// ---- tiling ----
#define BM   32          // tokens per block
#define HC   512         // H-chunk staged in LDS between the two GEMMs
#define HPAD (HC + 8)    // padded LDS row (bf16 elems) to spread banks

typedef __attribute__((ext_vector_type(16))) __bf16 v16bf;
typedef __attribute__((ext_vector_type(8)))  float  v8f;

// ---- workspace layout (PRE path) ----
#define GATE_BYTES ((size_t)BTOK * NEXP * 4)                 // 128 KB
#define XSW_BYTES  ((size_t)BTOK * DDIM * 2)                 // 8 MB
#define W1SW_BYTES ((size_t)NEXP * HDIM * DDIM * 2)          // 64 MB
#define W2SW_BYTES ((size_t)NEXP * ODIM * HDIM * 2)          // 64 MB
#define WS_NEED    (GATE_BYTES + XSW_BYTES + W1SW_BYTES + W2SW_BYTES)

static __device__ inline v8f wmma_bf16(v16bf a, v16bf b, v8f c) {
  // 8 args: (neg_a, A, neg_b, B, c_mod, C, reuse_a, reuse_b)
  return __builtin_amdgcn_wmma_f32_16x16x32_bf16(false, a, false, b, (short)0, c,
                                                 false, false);
}

// Load a pre-swizzled 32-byte fragment slice (16 bf16) for this lane.
union frag16u { uint4 q[2]; v16bf v; };
static __device__ inline v16bf load_frag_sw(const __bf16* __restrict__ p) {
  frag16u f;
  const uint4* q = (const uint4*)p;
  f.q[0] = q[0];
  f.q[1] = q[1];
  return f.v;
}

// ---- fallback (f32-direct) fragment loads, ISA 7.12.2 layouts ----
// A (16x32 bf16): lane-half 0 holds K {0..7,16..23}, half 1 holds K {8..15,24..31}.
static __device__ inline v16bf load_a_f32(const float* __restrict__ row, int lhalf) {
  v16bf r;
  const float* p0 = row + lhalf * 8;
  const float* p1 = row + 16 + lhalf * 8;
#pragma unroll
  for (int i = 0; i < 8; ++i) {
    r[i]     = (__bf16)p0[i];
    r[i + 8] = (__bf16)p1[i];
  }
  return r;
}
static __device__ inline v16bf load_a_lds(const __bf16* row, int lhalf) {
  v16bf r;
  const __bf16* p0 = row + lhalf * 8;
  const __bf16* p1 = row + 16 + lhalf * 8;
#pragma unroll
  for (int i = 0; i < 8; ++i) {
    r[i]     = p0[i];
    r[i + 8] = p1[i];
  }
  return r;
}
// B (32x16 bf16): lane&15 = output column; lane-half 0 holds K {0..7,16..23}, half 1 holds K {8..15,24..31}.
static __device__ inline v16bf load_b_f32(const float* __restrict__ colrow, int lhalf) {
  v16bf r;
  const float* p0 = colrow + lhalf * 8;
  const float* p1 = colrow + 16 + lhalf * 8;
#pragma unroll
  for (int i = 0; i < 8; ++i) {
    r[i]     = (__bf16)p0[i];
    r[i + 8] = (__bf16)p1[i];
  }
  return r;
}

// ---------------------------------------------------------------------------
// Swizzle pass: f32 [N][R][K] row-major -> bf16 B-fragment layout
// [N][R/16 coltile][K/32 kstep][lane 0..31][16], lane-half 0 holds K {0..7,16..23}, half 1 holds K {8..15,24..31}.
// One thread per 16-element K-run: 64B coalesced read, two 16B writes 512B apart.
// ---------------------------------------------------------------------------
__global__ __launch_bounds__(256) void swizzle_w_kernel(
    const float* __restrict__ src, __bf16* __restrict__ dst, int R, int K)
{
  const size_t g = (size_t)blockIdx.x * 256 + threadIdx.x;
  const int k16cnt = K >> 4;
  const int k16 = (int)(g % k16cnt);
  const size_t t = g / k16cnt;
  const int r = (int)(t % R);
  const int n = (int)(t / R);
  const float* s = src + ((size_t)n * R + r) * K + (size_t)k16 * 16;
  const int ct = r >> 4, lcol = r & 15, ks = k16 >> 1, lh = k16 & 1;
  const int lane = lcol;
  __bf16* d = dst + ((((size_t)n * (R >> 4) + ct) * (K >> 5) + ks) * 32 + lane) * 16;
#pragma unroll
  for (int i = 0; i < 16; ++i) d[(i >> 3) * 256 + lh * 8 + (i & 7)] = (__bf16)s[i];
}

// ---------------------------------------------------------------------------
// Swizzle pass: x f32 [B][D] -> bf16 A-fragment layout
// [B/16 rowtile][D/32 kstep][lane][16] with per-lane K runs {0..7,16..23} /
// {8..15,24..31} baked in. One thread per 8-element run.
// ---------------------------------------------------------------------------
__global__ __launch_bounds__(256) void swizzle_x_kernel(
    const float* __restrict__ src, __bf16* __restrict__ dst)
{
  const size_t g = (size_t)blockIdx.x * 256 + threadIdx.x;
  const int d8cnt = DDIM >> 3;
  const int d8 = (int)(g % d8cnt);
  const int b = (int)(g / d8cnt);
  const int d0 = d8 * 8;
  const int ks = d0 >> 5;
  const int kin = d0 & 31;
  const int run = (kin < 16) ? 0 : 1;
  const int lh  = (run == 0) ? (kin >> 3) : ((kin - 16) >> 3);
  const int lane = lh * 16 + (b & 15);
  const int rt = b >> 4;
  const float* s = src + (size_t)b * DDIM + d0;
  __bf16* d = dst + ((((size_t)rt * (DDIM >> 5) + ks) * 32 + lane) * 16 + run * 8);
#pragma unroll
  for (int i = 0; i < 8; ++i) d[i] = (__bf16)s[i];
}

// ---------------------------------------------------------------------------
// Kernel 1: Boltzmann gate.  One wave32 per token.
// ---------------------------------------------------------------------------
__global__ __launch_bounds__(256) void gate_kernel(
    const float* __restrict__ x, const float* __restrict__ Wg,
    const float* __restrict__ bg, float* __restrict__ gate)
{
  const int wave = threadIdx.x >> 5;
  const int lane = threadIdx.x & 31;
  const int b = blockIdx.x * 8 + wave;

  float acc[NEXP];
#pragma unroll
  for (int n = 0; n < NEXP; ++n) acc[n] = 0.0f;

  const float* xb = x + (size_t)b * DDIM;
  for (int d = lane; d < DDIM; d += 32) {
    const float xv = xb[d];
#pragma unroll
    for (int n = 0; n < NEXP; ++n) acc[n] += xv * Wg[n * DDIM + d];
  }
#pragma unroll
  for (int n = 0; n < NEXP; ++n) {
#pragma unroll
    for (int off = 16; off >= 1; off >>= 1)
      acc[n] += __shfl_xor(acc[n], off, 32);
    acc[n] += bg[n];
  }

  const float invT = 0.36787944117144233f;  // 1/e
  float mx = acc[0];
#pragma unroll
  for (int n = 1; n < NEXP; ++n) mx = fmaxf(mx, acc[n]);
  float p[NEXP];
  float s = 0.0f;
#pragma unroll
  for (int n = 0; n < NEXP; ++n) { p[n] = __expf((acc[n] - mx) * invT); s += p[n]; }
  const float invs = 1.0f / s;
#pragma unroll
  for (int n = 0; n < NEXP; ++n) p[n] *= invs;

  float keep[NEXP];
  float ksum = 0.0f;
#pragma unroll
  for (int n = 0; n < NEXP; ++n) {
    int rank = 0;
#pragma unroll
    for (int m = 0; m < NEXP; ++m)
      rank += (acc[m] > acc[n]) || (acc[m] == acc[n] && m < n);
    keep[n] = (rank < NACT) ? p[n] : 0.0f;
    ksum += keep[n];
  }
  const float inv = 1.0f / (ksum + 1e-8f);
  if (lane < NEXP) gate[b * NEXP + lane] = keep[lane] * inv;
}

// ---------------------------------------------------------------------------
// Kernel 2: fused dense-all-expert FFN, bf16 WMMA, f32 accumulate.
// Layer-1 uses 2x2 register blocking per wave (32x32 patch): 2 A-frags +
// 2 B-frags feed 4 WMMAs per K-step -> 1 KB L2 traffic per WMMA.
// PRE=true : fragments streamed pre-swizzled bf16 from ws (coalesced, no cvt).
// PRE=false: direct f32 global loads with on-the-fly cvt.
// ---------------------------------------------------------------------------
template <bool PRE>
__global__ __launch_bounds__(512, 1) void moe_ffn_kernel(
    const float* __restrict__ x,  const float* __restrict__ W1,
    const float* __restrict__ b1, const float* __restrict__ W2,
    const float* __restrict__ b2, const float* __restrict__ gate,
    const __bf16* __restrict__ xsw, const __bf16* __restrict__ w1sw,
    const __bf16* __restrict__ w2sw, float* __restrict__ out)
{
  __shared__ __bf16 hbuf[BM * HPAD];   // relu(x@W1^T+b1) chunk, bf16 (~33 KB)
  __shared__ float  wsh[BM * NEXP];    // gate weights for this token block

  const int tid   = threadIdx.x;
  const int wave  = tid >> 5;          // 0..15
  const int lane  = tid & 31;
  const int lhalf = lane >> 4;
  const int lmod  = lane & 15;
  const int b0    = blockIdx.x * BM;

  const int l2_cg = wave;              // layer-2: O-col-tiles {wave*4+j}

  if (tid < BM * NEXP) wsh[tid] = gate[b0 * NEXP + tid];

  v8f outacc[8];
#pragma unroll
  for (int t = 0; t < 8; ++t)
#pragma unroll
    for (int i = 0; i < 8; ++i) outacc[t][i] = 0.0f;

  // PRE: per-wave A-fragment bases for both M-tiles (lane-major 512-bf16 tiles)
  const __bf16* xfragA[2] = {
      PRE ? xsw + (((size_t)(blockIdx.x * 2 + 0) * (DDIM >> 5)) * 32 + lane) * 16
          : nullptr,
      PRE ? xsw + (((size_t)(blockIdx.x * 2 + 1) * (DDIM >> 5)) * 32 + lane) * 16
          : nullptr};

  for (int n = 0; n < NEXP; ++n) {
    v8f eacc[8];
#pragma unroll
    for (int t = 0; t < 8; ++t)
#pragma unroll
      for (int i = 0; i < 8; ++i) eacc[t][i] = 0.0f;

    for (int hc = 0; hc < HDIM / HC; ++hc) {
      const int hbase = hc * HC;

      __syncthreads();  // hbuf from previous chunk fully consumed

      // ---- layer 1: 32x32 patch per wave (2 M-tiles x 2 H-col-tiles) ----
      v8f acc1[4];
#pragma unroll
      for (int t = 0; t < 4; ++t)
#pragma unroll
        for (int i = 0; i < 8; ++i) acc1[t][i] = 0.0f;

      const int ct0 = 2 * wave;                    // chunk-local col-tile base
      const int hct0 = (hbase >> 4) + ct0;         // global h col-tile base

      if (PRE) {
        const __bf16* w1frag0 =
            w1sw + ((((size_t)n * (HDIM >> 4) + hct0) * (DDIM >> 5)) * 32 + lane) * 16;
        const __bf16* w1frag1 = w1frag0 + (size_t)(DDIM >> 5) * 512;
#pragma unroll 2
        for (int ks = 0; ks < (DDIM >> 5); ++ks) {
          v16bf a0 = load_frag_sw(xfragA[0] + (size_t)ks * 512);
          v16bf a1 = load_frag_sw(xfragA[1] + (size_t)ks * 512);
          v16bf bb0 = load_frag_sw(w1frag0 + (size_t)ks * 512);
          v16bf bb1 = load_frag_sw(w1frag1 + (size_t)ks * 512);
          acc1[0] = wmma_bf16(a0, bb0, acc1[0]);
          acc1[1] = wmma_bf16(a1, bb0, acc1[1]);
          acc1[2] = wmma_bf16(a0, bb1, acc1[2]);
          acc1[3] = wmma_bf16(a1, bb1, acc1[3]);
        }
      } else {
        const float* aptr0 = x + (size_t)(b0 + 0 * 16 + lmod) * DDIM;
        const float* aptr1 = x + (size_t)(b0 + 1 * 16 + lmod) * DDIM;
        const float* bptr0 = W1 + ((size_t)n * HDIM + hct0 * 16 + lmod) * DDIM;
        const float* bptr1 = bptr0 + (size_t)16 * DDIM;
#pragma unroll 2
        for (int k = 0; k < DDIM; k += 32) {
          v16bf a0 = load_a_f32(aptr0 + k, lhalf);
          v16bf a1 = load_a_f32(aptr1 + k, lhalf);
          v16bf bb0 = load_b_f32(bptr0 + k, lhalf);
          v16bf bb1 = load_b_f32(bptr1 + k, lhalf);
          acc1[0] = wmma_bf16(a0, bb0, acc1[0]);
          acc1[1] = wmma_bf16(a1, bb0, acc1[1]);
          acc1[2] = wmma_bf16(a0, bb1, acc1[2]);
          acc1[3] = wmma_bf16(a1, bb1, acc1[3]);
        }
      }

      // bias + relu + stage to LDS (C/D layout: VGPR i -> row i + 8*lhalf)
#pragma unroll
      for (int ctl = 0; ctl < 2; ++ctl) {
        const float bias1 = b1[n * HDIM + (hct0 + ctl) * 16 + lmod];
        const int   ccol  = (ct0 + ctl) * 16 + lmod;  // chunk-local column
#pragma unroll
        for (int mt = 0; mt < 2; ++mt) {
          const v8f& a = acc1[ctl * 2 + mt];
#pragma unroll
          for (int i = 0; i < 8; ++i) {
            const int r = mt * 16 + i + lhalf * 8;
            float v = a[i] + bias1;
            v = v > 0.0f ? v : 0.0f;
            hbuf[r * HPAD + ccol] = (__bf16)v;
          }
        }
      }
      __syncthreads();

      // ---- layer 2: accumulate this chunk into per-expert tiles ----
#pragma unroll 4
      for (int k = 0; k < HC; k += 32) {
        v16bf a0 = load_a_lds(&hbuf[(0 * 16 + lmod) * HPAD + k], lhalf);
        v16bf a1 = load_a_lds(&hbuf[(1 * 16 + lmod) * HPAD + k], lhalf);
        const int ksg = (hbase + k) >> 5;            // global kstep in H
#pragma unroll
        for (int j = 0; j < 4; ++j) {
          const int otile = l2_cg * 4 + j;
          v16bf bfrag;
          if (PRE) {
            bfrag = load_frag_sw(
                w2sw + ((((size_t)n * (ODIM >> 4) + otile) * (HDIM >> 5) + ksg) * 32 +
                        lane) * 16);
          } else {
            const int ocol = otile * 16 + lmod;
            bfrag = load_b_f32(W2 + ((size_t)n * ODIM + ocol) * HDIM + hbase + k,
                               lhalf);
          }
          eacc[j * 2 + 0] = wmma_bf16(a0, bfrag, eacc[j * 2 + 0]);
          eacc[j * 2 + 1] = wmma_bf16(a1, bfrag, eacc[j * 2 + 1]);
        }
      }
    }

    // fold expert: out += w[b,n] * (eacc + b2[n,:])
#pragma unroll
    for (int j = 0; j < 4; ++j) {
      const int ocol    = (l2_cg * 4 + j) * 16 + lmod;
      const float bias2 = b2[n * ODIM + ocol];
#pragma unroll
      for (int mi = 0; mi < 2; ++mi) {
#pragma unroll
        for (int i = 0; i < 8; ++i) {
          const int r = mi * 16 + i + lhalf * 8;
          const float wgt = wsh[r * NEXP + n];
          outacc[j * 2 + mi][i] += wgt * (eacc[j * 2 + mi][i] + bias2);
        }
      }
    }
  }

  // final store
#pragma unroll
  for (int j = 0; j < 4; ++j) {
    const int ocol = (l2_cg * 4 + j) * 16 + lmod;
#pragma unroll
    for (int mi = 0; mi < 2; ++mi) {
#pragma unroll
      for (int i = 0; i < 8; ++i) {
        const int r = b0 + mi * 16 + i + lhalf * 8;
        out[(size_t)r * ODIM + ocol] = outacc[j * 2 + mi][i];
      }
    }
  }
}

// ---------------------------------------------------------------------------
extern "C" void kernel_launch(void* const* d_in, const int* in_sizes, int n_in,
                              void* d_out, int out_size, void* d_ws, size_t ws_size,
                              hipStream_t stream) {
  (void)in_sizes; (void)n_in; (void)out_size;
  const float* x  = (const float*)d_in[0];
  const float* W1 = (const float*)d_in[1];
  const float* b1 = (const float*)d_in[2];
  const float* W2 = (const float*)d_in[3];
  const float* b2 = (const float*)d_in[4];
  const float* Wg = (const float*)d_in[5];
  const float* bg = (const float*)d_in[6];
  float* out   = (float*)d_out;

  char*  ws    = (char*)d_ws;
  float* gatew = (float*)ws;                               // [BTOK, NEXP]

  gate_kernel<<<BTOK / 8, 256, 0, stream>>>(x, Wg, bg, gatew);

  if (ws_size >= WS_NEED) {
    __bf16* xsw  = (__bf16*)(ws + GATE_BYTES);
    __bf16* w1sw = (__bf16*)(ws + GATE_BYTES + XSW_BYTES);
    __bf16* w2sw = (__bf16*)(ws + GATE_BYTES + XSW_BYTES + W1SW_BYTES);

    swizzle_x_kernel<<<(BTOK * DDIM / 8) / 256, 256, 0, stream>>>(x, xsw);
    swizzle_w_kernel<<<(int)((size_t)NEXP * HDIM * DDIM / 16 / 256), 256, 0,
                      stream>>>(W1, w1sw, HDIM, DDIM);
    swizzle_w_kernel<<<(int)((size_t)NEXP * ODIM * HDIM / 16 / 256), 256, 0,
                      stream>>>(W2, w2sw, ODIM, HDIM);

    moe_ffn_kernel<true><<<BTOK / BM, 512, 0, stream>>>(
        x, W1, b1, W2, b2, gatew, xsw, w1sw, w2sw, out);
  } else {
    moe_ffn_kernel<false><<<BTOK / BM, 512, 0, stream>>>(
        x, W1, b1, W2, b2, gatew, nullptr, nullptr, nullptr, out);
  }
}
